// Attention_89936615178969
// MI455X (gfx1250) — matrix operand order repaired and checked by reading, not compiled, not run on hardware
//
#include <hip/hip_runtime.h>
#include <hip/hip_bf16.h>

// ---------------------------------------------------------------------------
// MI455X (gfx1250) attention pipeline, bf16 WMMA (V_WMMA_F32_16X16X32_BF16)
//   x[B,N,C] -> qkv GEMM -> per-head LN(q,k) -> flash attention -> proj GEMM
//   B=2, N=2048, C=1024, H=16, D=64
// LDS staging via GLOBAL_LOAD_ASYNC_TO_LDS_B128 (ASYNCcnt), double-buffered so
// DMA overlaps the WMMA stream; one barrier per K-step.
// ---------------------------------------------------------------------------

typedef unsigned short u16;
typedef unsigned int   u32;
typedef __attribute__((ext_vector_type(16))) u16    v16u;
typedef __attribute__((ext_vector_type(8)))  u16    v8u;
typedef __attribute__((ext_vector_type(16))) __bf16 v16bf;
typedef __attribute__((ext_vector_type(8)))  float  v8f;

union FragU { v16u v; v8u h[2]; };

__device__ __forceinline__ u16 f2bf(float f) {
  u32 x = __builtin_bit_cast(u32, f);
  u32 r = x + 0x7FFFu + ((x >> 16) & 1u);   // round-to-nearest-even
  return (u16)(r >> 16);
}

__device__ __forceinline__ v8f wmma_bf16(v16u a, v16u b, v8f c) {
  return __builtin_amdgcn_wmma_f32_16x16x32_bf16(
      false, __builtin_bit_cast(v16bf, a),
      false, __builtin_bit_cast(v16bf, b),
      (short)0, c, false, false);
}

// ---- async global->LDS 16B copy (gfx1250 ASYNCcnt path when available) ----
#if defined(__has_builtin)
#if __has_builtin(__builtin_amdgcn_global_load_async_to_lds_b128)
#define HAVE_ASYNC_LDS 1
#endif
#endif

#ifdef HAVE_ASYNC_LDS
typedef int v4i __attribute__((vector_size(4 * sizeof(int))));
typedef __attribute__((address_space(1))) v4i* as1_v4i_p;   // global
typedef __attribute__((address_space(3))) v4i* as3_v4i_p;   // LDS
__device__ __forceinline__ void cp16(const u16* g, u16* l) {
  __builtin_amdgcn_global_load_async_to_lds_b128(
      (as1_v4i_p)g, (as3_v4i_p)l, 0, 0);
}
__device__ __forceinline__ void cp16_wait() {
#if __has_builtin(__builtin_amdgcn_s_wait_asynccnt)
  __builtin_amdgcn_s_wait_asynccnt(0);
#else
  asm volatile("s_wait_asynccnt 0" ::: "memory");
#endif
}
#else
__device__ __forceinline__ void cp16(const u16* g, u16* l) {
  *(v8u*)l = *(const v8u*)g;
}
__device__ __forceinline__ void cp16_wait() {}
#endif

// ---------------------------------------------------------------------------
// f32 -> bf16 conversion (grid-stride)
// ---------------------------------------------------------------------------
__global__ void cvt_bf16(const float* __restrict__ in, u16* __restrict__ out, int n) {
  int i = blockIdx.x * blockDim.x + threadIdx.x;
  int stride = gridDim.x * blockDim.x;
  for (; i < n; i += stride) out[i] = f2bf(in[i]);
}

// f32 [K][N] -> bf16 transposed [N][K]  (one-time; output writes coalesced)
__global__ void cvt_bf16_t(const float* __restrict__ in, u16* __restrict__ out,
                           int N, int K) {
  int i = blockIdx.x * blockDim.x + threadIdx.x;
  int stride = gridDim.x * blockDim.x;
  int total = N * K;
  for (; i < total; i += stride) {
    int n = i / K, k = i - n * K;
    out[i] = f2bf(in[(size_t)k * N + n]);
  }
}

// ---------------------------------------------------------------------------
// bf16 WMMA GEMM: C[M,N](f32) = A[M,K](bf16,row) * BT[N,K](bf16, pre-transposed)
//                 + bias[N]
// Block 256 threads = 8 waves; tile BM=64, BN=128, BK=64; double-buffered LDS.
// Wave grid 2x4, each wave -> 32x32 output; 8 WMMA per wave per K-step.
// ---------------------------------------------------------------------------
__global__ __launch_bounds__(256) void gemm_bf16(
    const u16* __restrict__ A, const u16* __restrict__ BT,
    const float* __restrict__ bias, float* __restrict__ C,
    int M, int N, int K) {
  __shared__ __align__(16) u16 As[2][64 * 64];    // 2 x 8 KB
  __shared__ __align__(16) u16 Bt[2][128 * 64];   // 2 x 16 KB, Bt[n][k]

  const int tid  = threadIdx.x;
  const int lane = tid & 31;
  const int w    = tid >> 5;
  const int wm   = w & 1;        // 0..1  (row of wave grid)
  const int wn   = w >> 1;       // 0..3  (col of wave grid)
  const int m0   = blockIdx.y * 64;
  const int n0   = blockIdx.x * 128;
  const int half = (lane >= 16);
  const int l16  = lane & 15;
  (void)M;

  auto stage = [&](int kb, int sb) {
    // A tile 64x64 = 512 chunks of 8 u16 (2 per thread)
#pragma unroll
    for (int i = 0; i < 2; ++i) {
      int idx = tid + i * 256;
      int r = idx >> 3, c8 = (idx & 7) * 8;
      cp16(&A[(size_t)(m0 + r) * K + kb + c8], &As[sb][r * 64 + c8]);
    }
    // Bt tile 128x64 = 1024 chunks (4 per thread)
#pragma unroll
    for (int i = 0; i < 4; ++i) {
      int idx = tid + i * 256;
      int n = idx >> 3, c8 = (idx & 7) * 8;
      cp16(&BT[(size_t)(n0 + n) * K + kb + c8], &Bt[sb][n * 64 + c8]);
    }
  };

  v8f acc[2][2] = {};

  stage(0, 0);
  cp16_wait();
  __syncthreads();

  int buf = 0;
  for (int kb = 0; kb < K; kb += 64, buf ^= 1) {
    if (kb + 64 < K) stage(kb + 64, buf ^ 1);   // prefetch overlaps compute

#pragma unroll
    for (int kc = 0; kc < 2; ++kc) {
      FragU af[2], bf[2];
#pragma unroll
      for (int i = 0; i < 2; ++i) {
        int row = wm * 32 + i * 16 + l16;
        int bse = kc * 32 + (half ? 8 : 0);
        af[i].h[0] = *(const v8u*)&As[buf][row * 64 + bse];
        af[i].h[1] = *(const v8u*)&As[buf][row * 64 + bse + 16];
      }
#pragma unroll
      for (int j = 0; j < 2; ++j) {
        int col = wn * 32 + j * 16 + l16;
        int bse = kc * 32 + (half ? 8 : 0);
        bf[j].h[0] = *(const v8u*)&Bt[buf][col * 64 + bse];
        bf[j].h[1] = *(const v8u*)&Bt[buf][col * 64 + bse + 16];
      }
#pragma unroll
      for (int i = 0; i < 2; ++i)
#pragma unroll
        for (int j = 0; j < 2; ++j)
          acc[i][j] = wmma_bf16(af[i].v, bf[j].v, acc[i][j]);
    }

    cp16_wait();        // prefetch of kb+64 has landed
    __syncthreads();    // all waves done reading buf; buf^1 fully staged
  }

  // epilogue: C-layout lane(p, L): row = p + 8*(L>=16), col = L%16
#pragma unroll
  for (int i = 0; i < 2; ++i)
#pragma unroll
    for (int j = 0; j < 2; ++j)
#pragma unroll
      for (int p = 0; p < 8; ++p) {
        int row = m0 + wm * 32 + i * 16 + p + 8 * half;
        int col = n0 + wn * 32 + j * 16 + l16;
        C[(size_t)row * N + col] = acc[i][j][p] + bias[col];
      }
}

// ---------------------------------------------------------------------------
// pack_qkv: per-(token, head) wave applies LayerNorm over D=64 to q,k and
// writes bf16 q[B,H,N,D], k[B,H,N,D], and v transposed vT[B,H,D,N].
// One wave per (b,n,h): 65536 waves.
// ---------------------------------------------------------------------------
__global__ __launch_bounds__(256) void pack_qkv(
    const float* __restrict__ qkv,
    const float* __restrict__ q_scale, const float* __restrict__ q_bias,
    const float* __restrict__ k_scale, const float* __restrict__ k_bias,
    u16* __restrict__ qb, u16* __restrict__ kb, u16* __restrict__ vtb) {
  const int gw    = (blockIdx.x * 256 + threadIdx.x) >> 5;
  const int lane  = threadIdx.x & 31;
  const int h     = gw & 15;
  const int token = gw >> 4;        // b*2048 + n
  const int b     = token >> 11;
  const int n     = token & 2047;
  const int bh    = b * 16 + h;
  const size_t base = (size_t)token * 3072 + h * 64;

#pragma unroll
  for (int part = 0; part < 2; ++part) {   // 0 = q, 1 = k
    const float* sc = part ? k_scale : q_scale;
    const float* bi = part ? k_bias  : q_bias;
    size_t off = base + (size_t)part * 1024;
    float x0 = qkv[off + lane];
    float x1 = qkv[off + 32 + lane];
    float s = x0 + x1;
#pragma unroll
    for (int m = 1; m < 32; m <<= 1) s += __shfl_xor(s, m);
    float mean = s * (1.0f / 64.0f);
    float d0 = x0 - mean, d1 = x1 - mean;
    float v = d0 * d0 + d1 * d1;
#pragma unroll
    for (int m = 1; m < 32; m <<= 1) v += __shfl_xor(v, m);
    float rstd = rsqrtf(v * (1.0f / 64.0f) + 1e-6f);
    u16* dst = part ? kb : qb;
    size_t o = ((size_t)bh * 2048 + n) * 64;
    dst[o + lane]      = f2bf(d0 * rstd * sc[lane]      + bi[lane]);
    dst[o + 32 + lane] = f2bf(d1 * rstd * sc[lane + 32] + bi[lane + 32]);
  }
  // v: transpose into [B,H,D,N]
  {
    float v0 = qkv[base + 2048 + lane];
    float v1 = qkv[base + 2048 + 32 + lane];
    size_t ob = (size_t)bh * 64;
    vtb[(ob + lane)      * 2048 + n] = f2bf(v0);
    vtb[(ob + 32 + lane) * 2048 + n] = f2bf(v1);
  }
}

// ---------------------------------------------------------------------------
// Flash attention per (b,h): block = 64 query rows, 4 waves x 16 rows each.
// S = q k^T / 8 via WMMA, online softmax with per-lane row stats, P through
// wave-private LDS into A-fragment layout, O += P V via WMMA.
// K/V tiles double-buffered with async-to-LDS prefetch.
// ---------------------------------------------------------------------------
__global__ __launch_bounds__(128) void attn_fa(
    const u16* __restrict__ qb, const u16* __restrict__ kb,
    const u16* __restrict__ vtb, u16* __restrict__ aob) {
  const int qt   = blockIdx.x;          // query tile 0..31
  const int bh   = blockIdx.y;          // 0..31
  const int b    = bh >> 4, h = bh & 15;
  const int tid  = threadIdx.x;
  const int lane = tid & 31;
  const int w    = tid >> 5;
  const int half = (lane >= 16);
  const int l16  = lane & 15;

  __shared__ __align__(16) u16 Qs[64 * 64];         //  8 KB
  __shared__ __align__(16) u16 Ks[2][64 * 64];      // 16 KB
  __shared__ __align__(16) u16 VTs[2][64 * 64];     // 16 KB, [d][m]
  __shared__ __align__(16) u16 Ps[4][16 * 64];      //  8 KB wave-private

  const size_t hd = (size_t)bh * 2048 * 64;
  const u16* qh  = qb  + hd;
  const u16* kh  = kb  + hd;
  const u16* vth = vtb + hd;                        // [64][2048]

  auto stageKV = [&](int j, int sb) {
#pragma unroll
    for (int i = 0; i < 4; ++i) {
      int idx = tid + i * 128;
      int r = idx >> 3, c8 = (idx & 7) * 8;
      cp16(&kh[(size_t)(j * 64 + r) * 64 + c8], &Ks[sb][r * 64 + c8]);
      cp16(&vth[(size_t)r * 2048 + j * 64 + c8], &VTs[sb][r * 64 + c8]);
    }
  };

  // stage Q tile + first K/V block
#pragma unroll
  for (int i = 0; i < 4; ++i) {
    int idx = tid + i * 128;
    int r = idx >> 3, c8 = (idx & 7) * 8;
    cp16(&qh[(size_t)(qt * 64 + r) * 64 + c8], &Qs[r * 64 + c8]);
  }
  stageKV(0, 0);
  cp16_wait();
  __syncthreads();

  // q A-fragments (kept in registers for all key blocks)
  FragU qf[2];
#pragma unroll
  for (int kc = 0; kc < 2; ++kc) {
    int row = w * 16 + l16;
    int bse = kc * 32 + (half ? 8 : 0);
    qf[kc].h[0] = *(const v8u*)&Qs[row * 64 + bse];
    qf[kc].h[1] = *(const v8u*)&Qs[row * 64 + bse + 16];
  }

  v8f o[4] = {};
  float mst[8], lst[8];
#pragma unroll
  for (int p = 0; p < 8; ++p) { mst[p] = -3.0e38f; lst[p] = 0.0f; }

  int buf = 0;
  for (int j = 0; j < 2048 / 64; ++j, buf ^= 1) {
    if (j + 1 < 2048 / 64) stageKV(j + 1, buf ^ 1);   // prefetch overlaps

    // S = q k^T for this wave's 16 rows x 64 key columns
    v8f s[4];
#pragma unroll
    for (int t = 0; t < 4; ++t) {
      v8f st = {};
#pragma unroll
      for (int kc = 0; kc < 2; ++kc) {
        FragU bf;
        int row = t * 16 + l16;                 // key index = column of S
        int bse = kc * 32 + (half ? 8 : 0);
        bf.h[0] = *(const v8u*)&Ks[buf][row * 64 + bse];
        bf.h[1] = *(const v8u*)&Ks[buf][row * 64 + bse + 16];
        st = wmma_bf16(qf[kc].v, bf.v, st);
      }
      s[t] = st;
    }
#pragma unroll
    for (int t = 0; t < 4; ++t)
#pragma unroll
      for (int p = 0; p < 8; ++p) s[t][p] *= 0.125f;   // 1/sqrt(64)

    // online softmax update (row-wise; 16-lane group reductions)
#pragma unroll
    for (int p = 0; p < 8; ++p) {
      float rm = fmaxf(fmaxf(s[0][p], s[1][p]), fmaxf(s[2][p], s[3][p]));
#pragma unroll
      for (int m = 1; m < 16; m <<= 1) rm = fmaxf(rm, __shfl_xor(rm, m));
      float newm = fmaxf(mst[p], rm);
      float alpha = __expf(mst[p] - newm);
      float rs = 0.0f;
#pragma unroll
      for (int t = 0; t < 4; ++t) {
        s[t][p] = __expf(s[t][p] - newm);
        rs += s[t][p];
      }
#pragma unroll
      for (int m = 1; m < 16; m <<= 1) rs += __shfl_xor(rs, m);
      lst[p] = lst[p] * alpha + rs;
      mst[p] = newm;
#pragma unroll
      for (int t = 0; t < 4; ++t) o[t][p] *= alpha;
    }

    // P (C-layout) -> wave-private LDS -> A-fragment layout
#pragma unroll
    for (int t = 0; t < 4; ++t)
#pragma unroll
      for (int p = 0; p < 8; ++p)
        Ps[w][(p + 8 * half) * 64 + t * 16 + l16] = f2bf(s[t][p]);

    // O += P * V
#pragma unroll
    for (int kc = 0; kc < 2; ++kc) {
      FragU pf;
      int bse = kc * 32 + (half ? 8 : 0);
      pf.h[0] = *(const v8u*)&Ps[w][l16 * 64 + bse];
      pf.h[1] = *(const v8u*)&Ps[w][l16 * 64 + bse + 16];
#pragma unroll
      for (int t = 0; t < 4; ++t) {
        FragU vf;
        int row = t * 16 + l16;                 // output column d
        int bse2 = kc * 32 + (half ? 8 : 0);
        vf.h[0] = *(const v8u*)&VTs[buf][row * 64 + bse2];
        vf.h[1] = *(const v8u*)&VTs[buf][row * 64 + bse2 + 16];
        o[t] = wmma_bf16(pf.v, vf.v, o[t]);
      }
    }

    cp16_wait();        // prefetch of block j+1 has landed
    __syncthreads();    // all waves done with buf
  }

  // normalize and write bf16 attention output [token][C] with C = h*64 + d
#pragma unroll
  for (int t = 0; t < 4; ++t)
#pragma unroll
    for (int p = 0; p < 8; ++p) {
      int qrow  = qt * 64 + w * 16 + p + 8 * half;
      int token = b * 2048 + qrow;
      int col   = h * 64 + t * 16 + l16;
      aob[(size_t)token * 1024 + col] = f2bf(o[t][p] / lst[p]);
    }
}

// ---------------------------------------------------------------------------
// Host-side launch
// ---------------------------------------------------------------------------
extern "C" void kernel_launch(void* const* d_in, const int* in_sizes, int n_in,
                              void* d_out, int out_size, void* d_ws, size_t ws_size,
                              hipStream_t stream) {
  (void)in_sizes; (void)n_in; (void)out_size; (void)ws_size;
  const float* x       = (const float*)d_in[0];
  const float* w_qkv   = (const float*)d_in[1];
  const float* b_qkv   = (const float*)d_in[2];
  const float* q_scale = (const float*)d_in[3];
  const float* q_bias  = (const float*)d_in[4];
  const float* k_scale = (const float*)d_in[5];
  const float* k_bias  = (const float*)d_in[6];
  const float* w_proj  = (const float*)d_in[7];
  const float* b_proj  = (const float*)d_in[8];
  float* out = (float*)d_out;

  char* ws = (char*)d_ws;
  const size_t MB = 1024ull * 1024ull;
  u16*   x_b     = (u16*)(ws);              //  8 MB  bf16 x        [4096][1024]
  u16*   wqkv_t  = (u16*)(ws + 8 * MB);     //  6 MB  bf16 w_qkv^T  [3072][1024]
  u16*   wproj_t = (u16*)(ws + 14 * MB);    //  2 MB  bf16 w_proj^T [1024][1024]
  float* qkv     = (float*)(ws + 16 * MB);  // 48 MB  f32  qkv      [4096][3072]
  u16*   q_b     = (u16*)(ws + 64 * MB);    //  8 MB  bf16 q        [B,H,N,D]
  u16*   k_b     = (u16*)(ws + 72 * MB);    //  8 MB  bf16 k        [B,H,N,D]
  u16*   vt_b    = (u16*)(ws + 80 * MB);    //  8 MB  bf16 vT       [B,H,D,N]
  u16*   ao_b    = (u16*)(ws + 88 * MB);    //  8 MB  bf16 attnout  [4096][1024]

  cvt_bf16  <<<2048, 256, 0, stream>>>(x, x_b, 4096 * 1024);
  cvt_bf16_t<<<2048, 256, 0, stream>>>(w_qkv,  wqkv_t,  3072, 1024);
  cvt_bf16_t<<<1024, 256, 0, stream>>>(w_proj, wproj_t, 1024, 1024);

  // qkv = x @ w_qkv + b_qkv   (M=4096, N=3072, K=1024)
  gemm_bf16<<<dim3(3072 / 128, 4096 / 64), 256, 0, stream>>>(
      x_b, wqkv_t, b_qkv, qkv, 4096, 3072, 1024);

  // LayerNorm(q,k) + reshape to heads + bf16 + v transpose
  pack_qkv<<<(65536 * 32) / 256, 256, 0, stream>>>(
      qkv, q_scale, q_bias, k_scale, k_bias, q_b, k_b, vt_b);

  // flash attention per (b,h)
  attn_fa<<<dim3(32, 32), 128, 0, stream>>>(q_b, k_b, vt_b, ao_b);

  // out = attn @ w_proj + b_proj   (M=4096, N=1024, K=1024)
  gemm_bf16<<<dim3(1024 / 128, 4096 / 64), 256, 0, stream>>>(
      ao_b, wproj_t, b_proj, out, 4096, 1024, 1024);
}
